// SpatialAttentionModel_49254684950601
// MI455X (gfx1250) — matrix operand order repaired and checked by reading, not compiled, not run on hardware
//
#include <hip/hip_runtime.h>
#include <math.h>

typedef __attribute__((ext_vector_type(16))) _Float16 v16h;
typedef __attribute__((ext_vector_type(8)))  _Float16 v8h;
typedef __attribute__((ext_vector_type(8)))  float    v8f;

// ---------------------------------------------------------------------------
// WMMA helpers (CDNA5 gfx1250, wave32)
// ---------------------------------------------------------------------------
__device__ __forceinline__ v8f wmma_f32_f16(v16h a, v16h b, v8f c) {
  // D = A(16x32 f16) * B(32x16 f16) + C(16x16 f32)
  return __builtin_amdgcn_wmma_f32_16x16x32_f16(false, a, false, b, (short)0, c,
                                                false, false);
}

// A fragment: row-major f16 matrix; p = &A[row0 + (lane&15)][k0].
// Lane<16 holds K {0..7, 16..23}; lane>=16 holds K {8..15, 24..31}.
__device__ __forceinline__ v16h load_a16(const _Float16* p, int lane) {
  int kb = (lane >> 4) << 3;
  v8h lo = *(const v8h*)(p + kb);
  v8h hi = *(const v8h*)(p + 16 + kb);
  return __builtin_shufflevector(lo, hi, 0, 1, 2, 3, 4, 5, 6, 7, 8, 9, 10, 11,
                                 12, 13, 14, 15);
}

// B fragment for W^T: B[k][n] = W[n][k], W row-major f16 with leading dim ldw.
// Lanes 0-15: N=lane, K=k0+{0..7, 16..23} ; lanes 16-31: N=lane-16, K=k0+{8..15, 24..31}.
__device__ __forceinline__ v16h load_bt16(const _Float16* W, int ldw, int n0,
                                          int k0, int lane) {
  const _Float16* p = W + (size_t)(n0 + (lane & 15)) * ldw + k0 + ((lane >> 4) << 3);
  v8h lo = *(const v8h*)(p);
  v8h hi = *(const v8h*)(p + 16);
  return __builtin_shufflevector(lo, hi, 0, 1, 2, 3, 4, 5, 6, 7, 8, 9, 10, 11,
                                 12, 13, 14, 15);
}

__device__ __forceinline__ float sigm(float x) { return 1.0f / (1.0f + expf(-x)); }

// ---------------------------------------------------------------------------
// K0: f32 -> f16 weight conversion
// ---------------------------------------------------------------------------
__global__ void k_cvt(const float* __restrict__ s, _Float16* __restrict__ d, int n) {
  int i = blockIdx.x * blockDim.x + threadIdx.x;
  if (i < n) d[i] = (_Float16)s[i];
}

// ---------------------------------------------------------------------------
// K1: xW_all[s][b][j] = bih[j] + bhh[j] + gaze[b,s,:] . Wih[j,:]   (32*64*1024)
// ---------------------------------------------------------------------------
__global__ void k_xw(const float* __restrict__ gaze, const float* __restrict__ Wih,
                     const float* __restrict__ bih, const float* __restrict__ bhh,
                     float* __restrict__ xW) {
  int idx = blockIdx.x * blockDim.x + threadIdx.x;  // [s][b][j]
  int j = idx & 1023;
  int b = (idx >> 10) & 63;
  int s = idx >> 16;
  const float* g = gaze + (b * 32 + s) * 3;
  xW[idx] = bih[j] + bhh[j] + g[0] * Wih[j * 3 + 0] + g[1] * Wih[j * 3 + 1] +
            g[2] * Wih[j * 3 + 2];
}

// ---------------------------------------------------------------------------
// K2: initial h0/c0 via tiny MLPs on mean gaze (one-time, VALU)
// ---------------------------------------------------------------------------
__global__ void k_init(const float* __restrict__ gaze,
                       const float* __restrict__ Wh1, const float* __restrict__ bh1,
                       const float* __restrict__ Wh2, const float* __restrict__ bh2,
                       const float* __restrict__ Wc1, const float* __restrict__ bc1,
                       const float* __restrict__ Wc2, const float* __restrict__ bc2,
                       float* __restrict__ h0, float* __restrict__ c0) {
  extern __shared__ char smraw[];
  float* u  = (float*)smraw;            // 64*256 f32
  float* mg = (float*)(smraw + 65536);  // 64*3   f32
  int tid = threadIdx.x;
  for (int idx = tid; idx < 192; idx += 256) {
    int b = idx / 3, k = idx - b * 3;
    float s = 0.f;
    for (int t = 0; t < 32; ++t) s += gaze[(b * 32 + t) * 3 + k];
    mg[idx] = s * (1.0f / 32.0f);
  }
  __syncthreads();
  for (int pass = 0; pass < 2; ++pass) {
    const float* W1 = pass ? Wc1 : Wh1;
    const float* b1 = pass ? bc1 : bh1;
    const float* W2 = pass ? Wc2 : Wh2;
    const float* b2 = pass ? bc2 : bh2;
    float* out = pass ? c0 : h0;
    for (int idx = tid; idx < 16384; idx += 256) {
      int b = idx >> 8, j = idx & 255;
      float v = b1[j] + mg[b * 3 + 0] * W1[j * 3 + 0] +
                mg[b * 3 + 1] * W1[j * 3 + 1] + mg[b * 3 + 2] * W1[j * 3 + 2];
      u[idx] = tanhf(v);
    }
    __syncthreads();
    for (int idx = tid; idx < 16384; idx += 256) {
      int b = idx >> 8, j = idx & 255;
      float acc = b2[j];
      const float* ub = u + b * 256;
      const float* wr = W2 + j * 256;
      for (int k = 0; k < 256; ++k) acc += ub[k] * wr[k];
      out[idx] = tanhf(tanhf(acc));
    }
    __syncthreads();
  }
}

// ---------------------------------------------------------------------------
// K3: Vp_all[m][p] = cnn[m][:] @ W_cn^T + b_cn ; M=64*32*36, K=256, N=128
// 8 waves per block, one 16x16 tile per wave, K-loop of 8 WMMAs.
// ---------------------------------------------------------------------------
__global__ __launch_bounds__(256) void k_vp(const float* __restrict__ cnn,
                                            const _Float16* __restrict__ Wcn16,
                                            const float* __restrict__ bcn,
                                            float* __restrict__ Vp) {
  int lane = threadIdx.x & 31;
  int gw = blockIdx.x * 8 + (threadIdx.x >> 5);  // global wave-tile id
  int mt = gw >> 3, nt = gw & 7;
  int row0 = mt << 4, n0 = nt << 4;
  v8f acc;
  float bv = bcn[n0 + (lane & 15)];
#pragma unroll
  for (int r = 0; r < 8; ++r) acc[r] = bv;
  const float* arow = cnn + (size_t)(row0 + (lane & 15)) * 256;
  int kb = (lane >> 4) << 3;
#pragma unroll
  for (int kc = 0; kc < 8; ++kc) {
    int k0 = kc << 5;
    v16h a;
#pragma unroll
    for (int j = 0; j < 8; ++j) a[j] = (_Float16)arow[k0 + kb + j];
#pragma unroll
    for (int j = 0; j < 8; ++j) a[8 + j] = (_Float16)arow[k0 + 16 + kb + j];
    v16h b = load_bt16(Wcn16, 256, n0, k0, lane);
    acc = wmma_f32_f16(a, b, acc);
  }
#pragma unroll
  for (int r = 0; r < 8; ++r)
    Vp[(size_t)(row0 + r + ((lane >> 4) << 3)) * 128 + n0 + (lane & 15)] = acc[r];
}

// ---------------------------------------------------------------------------
// K4: persistent single-workgroup kernel: per-frame attention + LSTM + MLP.
// 1024 threads = 32 waves on one WGP; ~248KB dynamic LDS holds all state.
// ---------------------------------------------------------------------------
#define OFF_T16 65536
#define OFF_CF  131072
#define OFF_HL  196608
#define OFF_Z   229376
#define OFF_SW  238592
#define OFF_WW  247808
#define SMEM_MAIN 248320

__global__ __launch_bounds__(1024) void k_main(
    const float* __restrict__ cnn, const _Float16* __restrict__ Wlh16,
    const float* __restrict__ blh, const float* __restrict__ wwt,
    const _Float16* __restrict__ Whh16, const _Float16* __restrict__ Wm1h,
    const float* __restrict__ bm1, const _Float16* __restrict__ Wm2h,
    const float* __restrict__ bm2, const float* __restrict__ Wm3,
    const float* __restrict__ bm3, const float* __restrict__ h0,
    const float* __restrict__ c0, const float* __restrict__ Vp,
    const float* __restrict__ xW, float* __restrict__ out) {
  extern __shared__ char smraw[];
  _Float16* fc16 = (_Float16*)smraw;              // [64][512]; cols 256..511 = h (f16)
  _Float16* t16  = (_Float16*)(smraw + OFF_T16);  // [64][512] MLP temp
  float* cF  = (float*)(smraw + OFF_CF);          // [64][256] LSTM cell (f32)
  float* HlS = (float*)(smraw + OFF_HL);          // [64][128]
  float* zS  = (float*)(smraw + OFF_Z);           // [64][36]
  float* swS = (float*)(smraw + OFF_SW);          // [64][36]
  float* wwS = (float*)(smraw + OFF_WW);          // [128]

  int tid = threadIdx.x, lane = tid & 31, w = tid >> 5;  // 32 waves

  // ---- prologue: load h0/c0, w_wt ----
  for (int idx = tid; idx < 16384; idx += 1024) {
    int b = idx >> 8, j = idx & 255;
    fc16[b * 512 + 256 + j] = (_Float16)h0[idx];
    cF[idx] = c0[idx];
  }
  if (tid < 128) wwS[tid] = wwt[tid];
  __syncthreads();

  for (int frame = 0; frame < 32; ++frame) {
    // ---- 1) Hl = h @ W_lh^T + b_lh  (4x8 tiles, one per wave) ----
    {
      int mt = w >> 3, nt = w & 7;
      int row0 = mt << 4, n0 = nt << 4;
      v8f acc;
      float bv = blh[n0 + (lane & 15)];
#pragma unroll
      for (int r = 0; r < 8; ++r) acc[r] = bv;
      const _Float16* arow = fc16 + (size_t)(row0 + (lane & 15)) * 512 + 256;
#pragma unroll
      for (int kc = 0; kc < 8; ++kc) {
        v16h a = load_a16(arow + (kc << 5), lane);
        v16h b = load_bt16(Wlh16, 256, n0, kc << 5, lane);
        acc = wmma_f32_f16(a, b, acc);
      }
#pragma unroll
      for (int r = 0; r < 8; ++r)
        HlS[(row0 + r + ((lane >> 4) << 3)) * 128 + n0 + (lane & 15)] = acc[r];
    }
    __syncthreads();

    // ---- 2) z[b][g] = sum_p tanh(Hl[b][p] + Vp[b,frame,g,p]) * w_wt[p] ----
    for (int idx = tid; idx < 2304; idx += 1024) {
      int b = idx / 36, g = idx - b * 36;
      const float* vp = Vp + (size_t)((b * 32 + frame) * 36 + g) * 128;
      const float* hl = HlS + b * 128;
      float acc = 0.f;
      for (int p = 0; p < 128; ++p) acc += tanhf(hl[p] + vp[p]) * wwS[p];
      zS[idx] = acc;
    }
    __syncthreads();

    // ---- 3) scrambled softmax: zsc[b][g] = z[k%64][k/64], k=b*36+g ----
    if (tid < 64) {
      int b = tid;
      float vals[36];
      float mx = -1e30f;
#pragma unroll
      for (int g = 0; g < 36; ++g) {
        int k = b * 36 + g;
        float v = zS[(k & 63) * 36 + (k >> 6)];
        vals[g] = v;
        mx = fmaxf(mx, v);
      }
      float sum = 0.f;
#pragma unroll
      for (int g = 0; g < 36; ++g) {
        vals[g] = expf(vals[g] - mx);
        sum += vals[g];
      }
      float inv = 1.f / sum;
#pragma unroll
      for (int g = 0; g < 36; ++g) swS[b * 36 + g] = vals[g] * inv;
    }
    __syncthreads();

    // ---- 4) spatial_feat -> fc16[:, 0:256] ----
    for (int idx = tid; idx < 16384; idx += 1024) {
      int b = idx >> 8, f = idx & 255;
      const float* vb = cnn + (size_t)((b * 32 + frame) * 36) * 256 + f;
      const float* sw = swS + b * 36;
      float acc = 0.f;
#pragma unroll 4
      for (int g = 0; g < 36; ++g) acc += vb[g * 256] * sw[g];
      fc16[b * 512 + f] = (_Float16)acc;
    }
    __syncthreads();

    // ---- 5) LSTM: L steps, gates = xW[s] + h @ Whh^T ----
    int L = (frame == 0) ? 1 : frame;
    for (int s = 0; s < L; ++s) {
      const float* xWs = xW + (size_t)s * 64 * 1024;
      v8f acc[2][4];  // [combo half][gate i,f,g,o]
#pragma unroll
      for (int half = 0; half < 2; ++half) {
        int cidx = w + (half << 5);          // 0..63
        int rm = cidx >> 4, cn = cidx & 15;  // row tile 0..3, hidden-col tile 0..15
        int row0 = rm << 4;
#pragma unroll
        for (int q = 0; q < 4; ++q) {
          int n0 = (q << 8) + (cn << 4);
#pragma unroll
          for (int r = 0; r < 8; ++r)
            acc[half][q][r] =
                xWs[(size_t)(row0 + r + ((lane >> 4) << 3)) * 1024 + n0 + (lane & 15)];
        }
        const _Float16* arow = fc16 + (size_t)(row0 + (lane & 15)) * 512 + 256;
#pragma unroll
        for (int kc = 0; kc < 8; ++kc) {
          v16h a = load_a16(arow + (kc << 5), lane);
#pragma unroll
          for (int q = 0; q < 4; ++q) {
            v16h b = load_bt16(Whh16, 256, (q << 8) + (cn << 4), kc << 5, lane);
            acc[half][q] = wmma_f32_f16(a, b, acc[half][q]);
          }
        }
      }
      __syncthreads();  // all reads of h (fc16 upper half) complete
#pragma unroll
      for (int half = 0; half < 2; ++half) {
        int cidx = w + (half << 5);
        int rm = cidx >> 4, cn = cidx & 15;
        int row0 = rm << 4;
#pragma unroll
        for (int r = 0; r < 8; ++r) {
          int b = row0 + r + ((lane >> 4) << 3);
          int j = (cn << 4) + (lane & 15);
          float gi = acc[half][0][r], gf = acc[half][1][r];
          float gg = acc[half][2][r], go = acc[half][3][r];
          float cprev = cF[b * 256 + j];
          float cnew = sigm(gf) * cprev + sigm(gi) * tanhf(gg);
          float hnew = sigm(go) * tanhf(cnew);
          cF[b * 256 + j] = cnew;
          fc16[b * 512 + 256 + j] = (_Float16)hnew;
        }
      }
      __syncthreads();
    }

    // ---- 6a) t1 = fc @ Wm1^T + bm1 -> t16 ----
    {
      int mt = w >> 3;          // tile row constant per wave
      int ntb = (w & 7) << 2;   // 4 consecutive N tiles per wave
      int row0 = mt << 4;
      v8f acc[4];
#pragma unroll
      for (int jt = 0; jt < 4; ++jt) {
        float bv = bm1[((ntb + jt) << 4) + (lane & 15)];
#pragma unroll
        for (int r = 0; r < 8; ++r) acc[jt][r] = bv;
      }
      const _Float16* arow = fc16 + (size_t)(row0 + (lane & 15)) * 512;
#pragma unroll
      for (int kc = 0; kc < 16; ++kc) {
        v16h a = load_a16(arow + (kc << 5), lane);
#pragma unroll
        for (int jt = 0; jt < 4; ++jt) {
          v16h b = load_bt16(Wm1h, 512, (ntb + jt) << 4, kc << 5, lane);
          acc[jt] = wmma_f32_f16(a, b, acc[jt]);
        }
      }
#pragma unroll
      for (int jt = 0; jt < 4; ++jt)
#pragma unroll
        for (int r = 0; r < 8; ++r)
          t16[(size_t)(row0 + r + ((lane >> 4) << 3)) * 512 + ((ntb + jt) << 4) +
              (lane & 15)] = (_Float16)acc[jt][r];
    }
    __syncthreads();

    // ---- 6b) t2 = t1 @ Wm2^T + bm2 -> t16 (compute-all, barrier, store) ----
    {
      int mt = w >> 3;
      int ntb = (w & 7) << 2;
      int row0 = mt << 4;
      v8f acc[4];
#pragma unroll
      for (int jt = 0; jt < 4; ++jt) {
        float bv = bm2[((ntb + jt) << 4) + (lane & 15)];
#pragma unroll
        for (int r = 0; r < 8; ++r) acc[jt][r] = bv;
      }
      const _Float16* arow = t16 + (size_t)(row0 + (lane & 15)) * 512;
#pragma unroll
      for (int kc = 0; kc < 16; ++kc) {
        v16h a = load_a16(arow + (kc << 5), lane);
#pragma unroll
        for (int jt = 0; jt < 4; ++jt) {
          v16h b = load_bt16(Wm2h, 512, (ntb + jt) << 4, kc << 5, lane);
          acc[jt] = wmma_f32_f16(a, b, acc[jt]);
        }
      }
      __syncthreads();  // all reads of t1 done before overwrite
#pragma unroll
      for (int jt = 0; jt < 4; ++jt)
#pragma unroll
        for (int r = 0; r < 8; ++r)
          t16[(size_t)(row0 + r + ((lane >> 4) << 3)) * 512 + ((ntb + jt) << 4) +
              (lane & 15)] = (_Float16)acc[jt][r];
    }
    __syncthreads();

    // ---- 6c) pred = t2 @ Wm3^T + bm3 ; out[(frame*64+b)*6+n] ----
    if (tid < 384) {
      int b = tid / 6, n = tid - b * 6;
      const float* wr = Wm3 + n * 512;
      const _Float16* fr = t16 + b * 512;
      float acc = bm3[n];
      for (int k = 0; k < 512; ++k) acc += (float)fr[k] * wr[k];
      out[(frame * 64 + b) * 6 + n] = acc;
    }
    __syncthreads();
  }
}

// ---------------------------------------------------------------------------
// Host launcher
// ---------------------------------------------------------------------------
extern "C" void kernel_launch(void* const* d_in, const int* in_sizes, int n_in,
                              void* d_out, int out_size, void* d_ws, size_t ws_size,
                              hipStream_t stream) {
  const float* cnn  = (const float*)d_in[0];   // (64,32,36,256)
  const float* gaze = (const float*)d_in[1];   // (64,32,3)
  const float* W_lh = (const float*)d_in[2];   // (128,256)
  const float* b_lh = (const float*)d_in[3];
  const float* W_cn = (const float*)d_in[4];   // (128,256)
  const float* b_cn = (const float*)d_in[5];
  const float* w_wt = (const float*)d_in[6];   // (1,128)
  const float* Wih  = (const float*)d_in[7];   // (1024,3)
  const float* Whh  = (const float*)d_in[8];   // (1024,256)
  const float* bih  = (const float*)d_in[9];
  const float* bhh  = (const float*)d_in[10];
  const float* Wh1  = (const float*)d_in[11];
  const float* bh1  = (const float*)d_in[12];
  const float* Wh2  = (const float*)d_in[13];
  const float* bh2  = (const float*)d_in[14];
  const float* Wc1  = (const float*)d_in[15];
  const float* bc1  = (const float*)d_in[16];
  const float* Wc2  = (const float*)d_in[17];
  const float* bc2  = (const float*)d_in[18];
  const float* Wm1  = (const float*)d_in[19];  // (512,512)
  const float* bm1  = (const float*)d_in[20];
  const float* Wm2  = (const float*)d_in[21];  // (512,512)
  const float* bm2  = (const float*)d_in[22];
  const float* Wm3  = (const float*)d_in[23];  // (6,512)
  const float* bm3  = (const float*)d_in[24];

  char* ws = (char*)d_ws;
  size_t off = 0;
  float* Vp = (float*)(ws + off);      off += (size_t)64 * 32 * 36 * 128 * 4;  // 37.7MB
  float* xW = (float*)(ws + off);      off += (size_t)32 * 64 * 1024 * 4;      // 8.4MB
  float* h0 = (float*)(ws + off);      off += (size_t)64 * 256 * 4;
  float* c0 = (float*)(ws + off);      off += (size_t)64 * 256 * 4;
  _Float16* Wlh16 = (_Float16*)(ws + off); off += (size_t)128 * 256 * 2;
  _Float16* Wcn16 = (_Float16*)(ws + off); off += (size_t)128 * 256 * 2;
  _Float16* Whh16 = (_Float16*)(ws + off); off += (size_t)1024 * 256 * 2;
  _Float16* Wm1h  = (_Float16*)(ws + off); off += (size_t)512 * 512 * 2;
  _Float16* Wm2h  = (_Float16*)(ws + off); off += (size_t)512 * 512 * 2;

  // Stage 1: weight conversion f32 -> f16
  k_cvt<<<128, 256, 0, stream>>>(W_lh, Wlh16, 128 * 256);
  k_cvt<<<128, 256, 0, stream>>>(W_cn, Wcn16, 128 * 256);
  k_cvt<<<1024, 256, 0, stream>>>(Whh, Whh16, 1024 * 256);
  k_cvt<<<1024, 256, 0, stream>>>(Wm1, Wm1h, 512 * 512);
  k_cvt<<<1024, 256, 0, stream>>>(Wm2, Wm2h, 512 * 512);

  // Stage 2: LSTM input-gate constants and initial h/c
  k_xw<<<8192, 256, 0, stream>>>(gaze, Wih, bih, bhh, xW);
  k_init<<<1, 256, 66304, stream>>>(gaze, Wh1, bh1, Wh2, bh2, Wc1, bc1, Wc2, bc2,
                                    h0, c0);

  // Stage 3: bulk WMMA GEMM: Vp for all frames (73728 x 128, K=256)
  k_vp<<<4608, 256, 0, stream>>>(cnn, Wcn16, b_cn, Vp);

  // Stage 4: persistent single-WGP sequential model
  k_main<<<1, 1024, SMEM_MAIN, stream>>>(cnn, Wlh16, b_lh, w_wt, Whh16, Wm1h, bm1,
                                         Wm2h, bm2, Wm3, bm3, h0, c0, Vp, xW,
                                         (float*)d_out);
}
